// EGNN_39453569581176
// MI455X (gfx1250) — matrix operand order repaired and checked by reading, not compiled, not run on hardware
//
#include <hip/hip_runtime.h>
#include <hip/hip_bf16.h>
#include <stdint.h>

// ---------------------------------------------------------------------------
// CDNA5 (gfx1250) EGNN forward.
// Edge / node MLP GEMMs on v_wmma_f32_16x16x32_bf16 (fp32 accumulate).
// Edge kernel: 64 edges/WG, one B fragment reused across 4 edge sub-tiles.
// A-tile fill region selection is uniform per K-tile (scalar branches only).
// ---------------------------------------------------------------------------

typedef __attribute__((ext_vector_type(16))) __bf16 v16bf;
typedef __attribute__((ext_vector_type(8)))  float  v8f;

union Frag { uint32_t u[8]; v16bf v; };

__device__ __forceinline__ uint32_t bf16pair(float a, float b) {
    union { float f; uint32_t u; } ua, ub;
    ua.f = a; ub.f = b;
    uint32_t ra = (ua.u + 0x7FFFu + ((ua.u >> 16) & 1u)) >> 16;
    uint32_t rb = (ub.u + 0x7FFFu + ((ub.u >> 16) & 1u)) >> 16;
    return (ra & 0xFFFFu) | (rb << 16);
}

__device__ __forceinline__ float silu_f(float v) { return v / (1.f + expf(-v)); }

// ---------------------------------------------------------------------------
// Pack fp32 W[Kact, od] (row-major) into bf16 WMMA B-fragments.
// [ktile][ntile][lane][8 dwords]; lane L: n = ntile*16+(L&15), kb=(L>>4)*16,
// dword v -> K = kb+2v, kb+2v+1.
// ---------------------------------------------------------------------------
__global__ void pack_w_bf16(const float* __restrict__ W, uint32_t* __restrict__ out,
                            int Kact, int od, int nkt) {
    int nt  = od >> 4;
    int idx = blockIdx.x * blockDim.x + threadIdx.x;
    int total = nkt * nt * 32 * 8;
    if (idx >= total) return;
    int v    = idx & 7;
    int lane = (idx >> 3) & 31;
    int ct   = (idx >> 8) % nt;
    int kt   = (idx >> 8) / nt;
    int n  = ct * 16 + (lane & 15);
    int kb = (lane >> 4) * 16;
    int k0 = kt * 32 + kb + 2 * v;
    float a = (k0     < Kact) ? W[(size_t)k0 * od + n]       : 0.f;
    float b = (k0 + 1 < Kact) ? W[(size_t)(k0 + 1) * od + n] : 0.f;
    out[idx] = bf16pair(a, b);
}

// ---------------------------------------------------------------------------
// Edge kernel: 64 edges per workgroup, 8 waves (wave32).
//   edge_in = [h[dst] | h[src] | dist2 | edge_attr]   (K = 2*HD+301, padded)
//   m = silu(edge_in @ We + be); coeff = silu(m @ Wc + bc)
//   atomic segment-sum of m into agg_m[dst], rel*coeff into agg_c[dst].
// Wave w owns column tiles {w, w+8, ...} over all 4 edge sub-tiles, so each
// B fragment load feeds 4 WMMAs.
// ---------------------------------------------------------------------------
template<int HD, int OD>
__global__ __launch_bounds__(256)
void edge_kernel(const float* __restrict__ h, const float* __restrict__ x,
                 const int* __restrict__ srcI, const int* __restrict__ dstI,
                 const float* __restrict__ eattr,
                 const uint32_t* __restrict__ Wpack, const float* __restrict__ be,
                 const float* __restrict__ Wc, const float* __restrict__ bc,
                 float* __restrict__ agg_m, float* __restrict__ agg_c,
                 int E, int nkt)
{
    constexpr int NT    = OD / 16;   // output column tiles (16 or 8)
    constexpr int CPW   = NT / 8;    // column tiles per wave (2 or 1)
    constexpr int KT_HD = HD / 32;   // K-tiles fully inside one h region

    __shared__ uint32_t Alds[64 * 16];            // 64 rows x 32 bf16
    __shared__ float relx[64], rely[64], relz[64], d2s[64], coeffs[64];
    __shared__ int   dsts[64], srcs[64];

    const int tid  = threadIdx.x;
    const int lane = tid & 31;
    const int wave = tid >> 5;
    const int e0   = blockIdx.x * 64;

    if (tid < 64) {
        int e = e0 + tid;
        if (e < E) {
            int s = srcI[e], d = dstI[e];
            srcs[tid] = s; dsts[tid] = d;
            float rx = x[d * 3 + 0] - x[s * 3 + 0];
            float ry = x[d * 3 + 1] - x[s * 3 + 1];
            float rz = x[d * 3 + 2] - x[s * 3 + 2];
            relx[tid] = rx; rely[tid] = ry; relz[tid] = rz;
            d2s[tid] = rx * rx + ry * ry + rz * rz;
        } else {
            srcs[tid] = -1; dsts[tid] = -1;
            relx[tid] = rely[tid] = relz[tid] = 0.f;
            d2s[tid] = 0.f;
        }
        coeffs[tid] = 0.f;
    }
    __syncthreads();

    v8f acc[4][CPW];
    #pragma unroll
    for (int st = 0; st < 4; ++st)
        #pragma unroll
        for (int c = 0; c < CPW; ++c)
            acc[st][c] = v8f{0.f,0.f,0.f,0.f,0.f,0.f,0.f,0.f};

    const int m   = lane & 15;
    const int kbA = (lane >> 4) * 8;   // A dword base (0 or 8) -> K base 0 or 16
    const int fr  = tid >> 2;          // fill: row (0..63)
    const int fk  = (tid & 3) * 8;     // fill: k offset (0,8,16,24)

    for (int kt = 0; kt < nkt; ++kt) {
        // ---- cooperative fill: 64x32 bf16 tile; region chosen per-ktile ----
        {
            const int kg = kt * 32 + fk;
            const int d = dsts[fr], s = srcs[fr];
            float va[8];
            #pragma unroll
            for (int j = 0; j < 8; ++j) va[j] = 0.f;

            if (kt < KT_HD) {                         // fully h[dst]
                if (d >= 0) {
                    const float4* p = (const float4*)(h + (size_t)d * HD + kg);
                    float4 a0 = p[0], a1 = p[1];
                    va[0]=a0.x; va[1]=a0.y; va[2]=a0.z; va[3]=a0.w;
                    va[4]=a1.x; va[5]=a1.y; va[6]=a1.z; va[7]=a1.w;
                }
            } else if (kt < 2 * KT_HD) {              // fully h[src]
                if (d >= 0) {
                    const float4* p = (const float4*)(h + (size_t)s * HD + (kg - HD));
                    float4 a0 = p[0], a1 = p[1];
                    va[0]=a0.x; va[1]=a0.y; va[2]=a0.z; va[3]=a0.w;
                    va[4]=a1.x; va[5]=a1.y; va[6]=a1.z; va[7]=a1.w;
                }
            } else if ((kt > 2 * KT_HD) && ((kt * 32 - 2 * HD + 30) <= 299)) {
                // fully inside edge_attr (base misaligned by 1 float -> b32s)
                if (d >= 0) {
                    const float* ea = eattr + (size_t)(e0 + fr) * 300 + (kg - 2 * HD - 1);
                    #pragma unroll
                    for (int j = 0; j < 8; ++j) va[j] = ea[j];
                }
            } else {                                   // boundary / tail tile
                if (d >= 0) {
                    #pragma unroll
                    for (int j = 0; j < 8; ++j) {
                        int k = kg + j;
                        float vv = 0.f;
                        if (k == 2 * HD) vv = d2s[fr];
                        else if (k > 2 * HD && k < 2 * HD + 301)
                            vv = eattr[(size_t)(e0 + fr) * 300 + (k - 2 * HD - 1)];
                        va[j] = vv;
                    }
                }
            }
            uint4 pk;
            pk.x = bf16pair(va[0], va[1]);
            pk.y = bf16pair(va[2], va[3]);
            pk.z = bf16pair(va[4], va[5]);
            pk.w = bf16pair(va[6], va[7]);
            *(uint4*)&Alds[fr * 16 + (fk >> 1)] = pk;
        }
        __syncthreads();

        // ---- A fragments for the 4 edge sub-tiles ----
        Frag af[4];
        #pragma unroll
        for (int st = 0; st < 4; ++st) {
            const int rowA = st * 16 + m;
            #pragma unroll
            for (int q = 0; q < 8; ++q) {
                int dw = kbA + q;
                af[st].u[q] = Alds[rowA * 16 + dw];
            }
        }

        // ---- one B fragment per column tile, reused across 4 sub-tiles ----
        #pragma unroll
        for (int c = 0; c < CPW; ++c) {
            const int ct = wave + 8 * c;
            const uint32_t* bp = Wpack + ((size_t)(kt * NT + ct) * 32 + lane) * 8;
            Frag bfr;
            *(uint4*)(&bfr.u[0]) = *(const uint4*)(bp);
            *(uint4*)(&bfr.u[4]) = *(const uint4*)(bp + 4);
            #pragma unroll
            for (int st = 0; st < 4; ++st)
                acc[st][c] = __builtin_amdgcn_wmma_f32_16x16x32_bf16(
                    false, af[st].v, false, bfr.v, (short)0, acc[st][c], false, false);
        }
        __syncthreads();
    }

    // ---- epilogue: bias + SiLU, agg_m scatter, coeff dot reduction ----
    #pragma unroll
    for (int c = 0; c < CPW; ++c) {
        const int ct  = wave + 8 * c;
        const int col = ct * 16 + m;
        const float bias = be[col];
        const float wcv  = Wc[col];
        #pragma unroll
        for (int st = 0; st < 4; ++st) {
            #pragma unroll
            for (int r = 0; r < 8; ++r) {
                int erow = st * 16 + r + 8 * (lane >> 4);
                float pre = acc[st][c][r] + bias;
                float mm  = silu_f(pre);
                int d = dsts[erow];
                float p = mm * wcv;
                p += __shfl_xor(p, 1, 32);
                p += __shfl_xor(p, 2, 32);
                p += __shfl_xor(p, 4, 32);
                p += __shfl_xor(p, 8, 32);
                if (d >= 0) {
                    atomicAdd(&agg_m[(size_t)d * OD + col], mm);
                    if (m == 0) atomicAdd(&coeffs[erow], p);   // DS float atomic
                }
            }
        }
    }
    __syncthreads();

    if (tid < 64) {
        int d = dsts[tid];
        if (d >= 0) {
            float cval = coeffs[tid] + bc[0];
            float coeff = silu_f(cval);
            atomicAdd(&agg_c[d * 3 + 0], relx[tid] * coeff);
            atomicAdd(&agg_c[d * 3 + 1], rely[tid] * coeff);
            atomicAdd(&agg_c[d * 3 + 2], relz[tid] * coeff);
        }
    }
}

// ---------------------------------------------------------------------------
// Node kernel: h_raw = silu([h | agg_m] @ Wn + bn)   (K2 = HD+OD, 32-aligned,
// so each K-tile is entirely in one region -> uniform fill).
// ---------------------------------------------------------------------------
template<int HD, int OD>
__global__ __launch_bounds__(256)
void node_kernel(const float* __restrict__ h, const float* __restrict__ aggm,
                 const uint32_t* __restrict__ Wpack, const float* __restrict__ bnv,
                 float* __restrict__ hout, int N, int nkt)
{
    constexpr int NT    = OD / 16;
    constexpr int TPW   = NT / 4;
    constexpr int KT_HD = HD / 32;

    __shared__ uint32_t Alds[32 * 16];

    const int tid  = threadIdx.x;
    const int lane = tid & 31;
    const int wave = tid >> 5;
    const int n0   = blockIdx.x * 32;
    const int subtile = wave >> 2;
    const int cbase   = wave & 3;

    v8f acc[TPW];
    #pragma unroll
    for (int i = 0; i < TPW; ++i) acc[i] = v8f{0.f,0.f,0.f,0.f,0.f,0.f,0.f,0.f};

    const int m    = lane & 15;
    const int kbA  = (lane >> 4) * 8;
    const int rowA = subtile * 16 + m;
    const int fr   = tid >> 3;
    const int fk   = (tid & 7) * 4;

    for (int kt = 0; kt < nkt; ++kt) {
        {
            const int kg = kt * 32 + fk;
            const int node = n0 + fr;
            float4 a = {0.f, 0.f, 0.f, 0.f};
            if (node < N) {
                const float* base = (kt < KT_HD)
                    ? (h    + (size_t)node * HD + kg)
                    : (aggm + (size_t)node * OD + (kg - HD));
                a = *(const float4*)base;
            }
            Alds[fr * 16 + (fk >> 1)    ] = bf16pair(a.x, a.y);
            Alds[fr * 16 + (fk >> 1) + 1] = bf16pair(a.z, a.w);
        }
        __syncthreads();

        Frag af;
        #pragma unroll
        for (int q = 0; q < 8; ++q) {
            int dw = kbA + q;
            af.u[q] = Alds[rowA * 16 + dw];
        }

        #pragma unroll
        for (int i = 0; i < TPW; ++i) {
            int ct = cbase + 4 * i;
            const uint32_t* bp = Wpack + ((size_t)(kt * NT + ct) * 32 + lane) * 8;
            Frag bfr;
            *(uint4*)(&bfr.u[0]) = *(const uint4*)(bp);
            *(uint4*)(&bfr.u[4]) = *(const uint4*)(bp + 4);
            acc[i] = __builtin_amdgcn_wmma_f32_16x16x32_bf16(
                false, af.v, false, bfr.v, (short)0, acc[i], false, false);
        }
        __syncthreads();
    }

    #pragma unroll
    for (int i = 0; i < TPW; ++i) {
        int ct  = cbase + 4 * i;
        int col = ct * 16 + m;
        float bias = bnv[col];
        #pragma unroll
        for (int r = 0; r < 8; ++r) {
            int row  = subtile * 16 + r + 8 * (lane >> 4);
            int node = n0 + row;
            if (node < N)
                hout[(size_t)node * OD + col] = silu_f(acc[i][r] + bias);
        }
    }
}

// ---------------------------------------------------------------------------
__global__ void coord_update(const float* __restrict__ x, const float* __restrict__ aggc,
                             const float* __restrict__ cs, float* __restrict__ xout, int n3)
{
    int i = blockIdx.x * blockDim.x + threadIdx.x;
    if (i < n3) xout[i] = x[i] + cs[0] * aggc[i];
}

__global__ void bn_stats(const float* __restrict__ h, float* __restrict__ sums,
                         int N, int od, int rowsPer)
{
    int c = threadIdx.x;
    if (c >= od) return;
    int r0 = blockIdx.x * rowsPer;
    int r1 = r0 + rowsPer; if (r1 > N) r1 = N;
    float s = 0.f, s2 = 0.f;
    for (int r = r0; r < r1; ++r) {
        float v = h[(size_t)r * od + c];
        s += v; s2 += v * v;
    }
    atomicAdd(&sums[c], s);
    atomicAdd(&sums[256 + c], s2);
}

__global__ void bn_apply(const float* __restrict__ h, const float* __restrict__ sums,
                         const float* __restrict__ gamma, const float* __restrict__ beta,
                         float* __restrict__ out, int N, int od, int doSilu)
{
    long long i = (long long)blockIdx.x * blockDim.x + threadIdx.x;
    long long total = (long long)N * od;
    if (i >= total) return;
    int c = (int)(i % od);
    float invN = 1.f / (float)N;
    float mu  = sums[c] * invN;
    float var = sums[256 + c] * invN - mu * mu;
    float v = (h[i] - mu) * rsqrtf(var + 1e-5f) * gamma[c] + beta[c];
    if (doSilu) v = silu_f(v);
    out[i] = v;
}

// ---------------------------------------------------------------------------
// Host launch
// ---------------------------------------------------------------------------
extern "C" void kernel_launch(void* const* d_in, const int* in_sizes, int n_in,
                              void* d_out, int out_size, void* d_ws, size_t ws_size,
                              hipStream_t stream)
{
    const float* h0  = (const float*)d_in[0];
    const float* pos = (const float*)d_in[1];
    const int*   eix = (const int*)d_in[2];
    const float* eat = (const float*)d_in[3];
    const int N = in_sizes[0] / 256;
    const int E = in_sizes[2] / 2;
    const int* srcI = eix;
    const int* dstI = eix + E;

    static const int HDs[4] = {256, 256, 128, 128};
    static const int ODs[4] = {256, 128, 128, 128};

    char* ws = (char*)d_ws;
    size_t off = 0;
    auto walloc = [&](size_t bytes) -> void* {
        void* p = ws + off;
        off = (off + bytes + 255) & ~(size_t)255;
        return p;
    };

    float* bufA = (float*)walloc((size_t)N * 256 * 4);  // pre-BN (raw)
    float* bufB = (float*)walloc((size_t)N * 256 * 4);  // activation ping
    float* bufC = (float*)walloc((size_t)N * 128 * 4);  // activation pong
    float* aggm = (float*)walloc((size_t)N * 256 * 4);
    float* aggc = (float*)walloc((size_t)N * 3 * 4);
    float* x0   = (float*)walloc((size_t)N * 3 * 4);
    float* x1   = (float*)walloc((size_t)N * 3 * 4);
    float* bns  = (float*)walloc(512 * 4);

    uint32_t* wep[4]; uint32_t* wnp[4];
    int nktE[4], nktN[4];
    for (int li = 0; li < 4; ++li) {
        int hd = HDs[li], od = ODs[li];
        int kact = 2 * hd + 301;
        nktE[li] = (kact + 31) / 32;
        nktN[li] = (hd + od) / 32;
        wep[li] = (uint32_t*)walloc((size_t)nktE[li] * (od / 16) * 32 * 8 * 4);
        wnp[li] = (uint32_t*)walloc((size_t)nktN[li] * (od / 16) * 32 * 8 * 4);
    }

    // ---- pack weights (fp32 -> bf16 WMMA fragment order) ----
    for (int li = 0; li < 4; ++li) {
        int hd = HDs[li], od = ODs[li], base = 4 + li * 9;
        int kact = 2 * hd + 301;
        pack_w_bf16<<<nktE[li] * (od / 16), 256, 0, stream>>>(
            (const float*)d_in[base + 0], wep[li], kact, od, nktE[li]);
        pack_w_bf16<<<nktN[li] * (od / 16), 256, 0, stream>>>(
            (const float*)d_in[base + 4], wnp[li], hd + od, od, nktN[li]);
    }

    const float* hin = h0;
    const float* xin = pos;
    float* acts[2] = { bufB, bufC };
    float* xs[2]   = { x0, x1 };

    for (int li = 0; li < 4; ++li) {
        int od = ODs[li], base = 4 + li * 9;
        const float* be_  = (const float*)d_in[base + 1];
        const float* Wc_  = (const float*)d_in[base + 2];
        const float* bc_  = (const float*)d_in[base + 3];
        const float* bnv_ = (const float*)d_in[base + 5];
        const float* cs_  = (const float*)d_in[base + 6];
        const float* gam_ = (const float*)d_in[base + 7];
        const float* bet_ = (const float*)d_in[base + 8];

        hipMemsetAsync(aggm, 0, (size_t)N * od * 4, stream);
        hipMemsetAsync(aggc, 0, (size_t)N * 3 * 4, stream);
        hipMemsetAsync(bns, 0, 512 * 4, stream);

        int egrid = (E + 63) / 64;
        int ngrid = (N + 31) / 32;
        if (li == 0) {
            edge_kernel<256, 256><<<egrid, 256, 0, stream>>>(
                hin, xin, srcI, dstI, eat, wep[0], be_, Wc_, bc_, aggm, aggc, E, nktE[0]);
            node_kernel<256, 256><<<ngrid, 256, 0, stream>>>(
                hin, aggm, wnp[0], bnv_, bufA, N, nktN[0]);
        } else if (li == 1) {
            edge_kernel<256, 128><<<egrid, 256, 0, stream>>>(
                hin, xin, srcI, dstI, eat, wep[1], be_, Wc_, bc_, aggm, aggc, E, nktE[1]);
            node_kernel<256, 128><<<ngrid, 256, 0, stream>>>(
                hin, aggm, wnp[1], bnv_, bufA, N, nktN[1]);
        } else {
            edge_kernel<128, 128><<<egrid, 256, 0, stream>>>(
                hin, xin, srcI, dstI, eat, wep[li], be_, Wc_, bc_, aggm, aggc, E, nktE[li]);
            node_kernel<128, 128><<<ngrid, 256, 0, stream>>>(
                hin, aggm, wnp[li], bnv_, bufA, N, nktN[li]);
        }

        float* xout = xs[li & 1];
        coord_update<<<(N * 3 + 255) / 256, 256, 0, stream>>>(xin, aggc, cs_, xout, N * 3);

        const int rowsPer = 256;
        bn_stats<<<(N + rowsPer - 1) / rowsPer, 256, 0, stream>>>(bufA, bns, N, od, rowsPer);

        float* hact = (li == 3) ? (float*)d_out : acts[li & 1];
        long long tot = (long long)N * od;
        bn_apply<<<(unsigned)((tot + 255) / 256), 256, 0, stream>>>(
            bufA, bns, gam_, bet_, hact, N, od, (li < 3) ? 1 : 0);

        hin = hact;
        xin = xout;
    }
}
